// BinaryAttention_42752104464880
// MI455X (gfx1250) — 16-bit and 8-bit matrix operand order repaired and checked by reading, compiled, not run on hardware
//
#include <hip/hip_runtime.h>
#include <hip/hip_bf16.h>
#include <math.h>

typedef __attribute__((ext_vector_type(16))) _Float16 v16h;
typedef __attribute__((ext_vector_type(8)))  float    v8f;
typedef __attribute__((ext_vector_type(8)))  int      v8i;
typedef int v4i_vec __attribute__((vector_size(16)));

#if __has_builtin(__builtin_amdgcn_global_load_async_to_lds_b128) && \
    __has_builtin(__builtin_amdgcn_s_wait_asynccnt)
#define HAVE_ASYNC_LDS 1
typedef __attribute__((address_space(1))) v4i_vec gv4i;
typedef __attribute__((address_space(3))) v4i_vec lv4i;
#else
#define HAVE_ASYNC_LDS 0
#endif

#define BATCH   128
#define NTOK    197
#define CDIM    768
#define NHEADS  12
#define HDIM    64
#define ROWS    (BATCH*NTOK)
#define ROWT    788
#define MEANDIV (1.0f/(197.0f*64.0f))
#define SCALEQK 0.125f

__global__ __launch_bounds__(256)
void qkv_gemm_kernel(const float* __restrict__ x,
                     const float* __restrict__ w,
                     signed char* __restrict__ qsign,
                     signed char* __restrict__ ksign,
                     _Float16* __restrict__ vdeq,
                     float* __restrict__ qssum,
                     float* __restrict__ kssum) {
    __shared__ _Float16 As[2][32][40];
    __shared__ _Float16 Bs[2][64][40];
    __shared__ float    Cs[32 * 64];
    __shared__ float    rowstat[32];

    const int rt   = blockIdx.x;
    const int ct   = blockIdx.y;
    const int t    = threadIdx.x;
    const int wave = t >> 5, lane = t & 31;
    const int hi   = lane >> 4, ln = lane & 15;
    const int wr   = wave >> 2, wc = wave & 3;
    const int row0 = rt * 32;
    const int c0   = ct * 64;

    const int ar = t >> 3, acc4 = (t & 7) * 4;
    const int bc = t >> 2, bk8 = (t & 3) * 8;
    const float* aBase = x + (size_t)(row0 + ar) * CDIM + acc4;
    const float* bBase = w + (size_t)(c0 + bc) * CDIM + bk8;

    float4 aR, bR0, bR1;
    auto ldRegs = [&](int k0) {
        aR  = *(const float4*)(aBase + k0);
        bR0 = *(const float4*)(bBase + k0);
        bR1 = *(const float4*)(bBase + k0 + 4);
    };
    auto stLDS = [&](int bf) {
        As[bf][ar][acc4 + 0] = (_Float16)aR.x;
        As[bf][ar][acc4 + 1] = (_Float16)aR.y;
        As[bf][ar][acc4 + 2] = (_Float16)aR.z;
        As[bf][ar][acc4 + 3] = (_Float16)aR.w;
        Bs[bf][bc][bk8 + 0] = (_Float16)bR0.x;
        Bs[bf][bc][bk8 + 1] = (_Float16)bR0.y;
        Bs[bf][bc][bk8 + 2] = (_Float16)bR0.z;
        Bs[bf][bc][bk8 + 3] = (_Float16)bR0.w;
        Bs[bf][bc][bk8 + 4] = (_Float16)bR1.x;
        Bs[bf][bc][bk8 + 5] = (_Float16)bR1.y;
        Bs[bf][bc][bk8 + 6] = (_Float16)bR1.z;
        Bs[bf][bc][bk8 + 7] = (_Float16)bR1.w;
    };

    v8f acc;
#pragma unroll
    for (int j = 0; j < 8; ++j) acc[j] = 0.0f;

    ldRegs(0);
    stLDS(0);
    __syncthreads();

    int cur = 0;
    for (int i = 0; i < 24; ++i) {
        const int k0 = i * 32;
        if (i < 23) ldRegs(k0 + 32);
        if (i < 22) {
            __builtin_prefetch(aBase + k0 + 64, 0, 0);
            __builtin_prefetch(bBase + k0 + 64, 0, 0);
        }
        v16h af, bf;
#pragma unroll
        for (int e = 0; e < 16; ++e) {
            int pg = e >> 1, w2 = e & 1;
            int kk = (pg < 4 ? hi * 8 + 2 * pg : 16 + hi * 8 + 2 * (pg - 4)) + w2;
            af[e] = As[cur][wr * 16 + ln][kk];
        }
#pragma unroll
        for (int e = 0; e < 16; ++e)
            bf[e] = Bs[cur][wc * 16 + ln][(e < 8 ? hi * 8 + e : 16 + hi * 8 + (e - 8))];

        acc = __builtin_amdgcn_wmma_f32_16x16x32_f16(false, af, false, bf,
                                                     (short)0, acc, false, false);
        if (i < 23) stLDS(cur ^ 1);
        __syncthreads();
        cur ^= 1;
    }

#pragma unroll
    for (int j = 0; j < 8; ++j)
        Cs[(wr * 16 + hi * 8 + j) * 64 + wc * 16 + ln] = acc[j];
    __syncthreads();

    const int sec = ct / 12;
    const int h   = ct % 12;

    if (sec < 2) {
        float*       ssum = (sec == 0) ? qssum : kssum;
        signed char* sg   = (sec == 0) ? qsign : ksign;
        if (t < 32) {
            float s = 0.0f;
#pragma unroll 8
            for (int d = 0; d < 64; ++d) s += fabsf(Cs[t * 64 + d]);
            int R = row0 + t, b = R / NTOK;
            atomicAdd(&ssum[b * NHEADS + h], s);
        }
#pragma unroll
        for (int u = 0; u < 8; ++u) {
            int idx = t * 8 + u, rl = idx >> 6, d = idx & 63;
            int R = row0 + rl, b = R / NTOK, n = R % NTOK;
            sg[((size_t)(b * NHEADS + h) * NTOK + n) * 64 + d] =
                (Cs[idx] >= 0.0f) ? (signed char)1 : (signed char)-1;
        }
    } else {
        if (t < 32) {
            float m = 0.0f;
            for (int d = 0; d < 64; ++d) {
                float c = fminf(fmaxf(Cs[t * 64 + d], -2.0f), 2.0f);
                m = fmaxf(m, fabsf(c));
            }
            rowstat[t] = 127.0f / (m + 1e-8f);
        }
        __syncthreads();
#pragma unroll
        for (int u = 0; u < 8; ++u) {
            int idx = t * 8 + u, rl = idx >> 6, d = idx & 63;
            int R = row0 + rl, b = R / NTOK, n = R % NTOK;
            float c = fminf(fmaxf(Cs[idx], -2.0f), 2.0f);
            float s = rowstat[rl];
            vdeq[((size_t)(b * NHEADS + h) * NTOK + n) * 64 + d] =
                (_Float16)(rintf(c * s) / s);
        }
    }
}

__global__ __launch_bounds__(128)
void attn_kernel(const signed char* __restrict__ qsign,
                 const signed char* __restrict__ ksign,
                 const _Float16* __restrict__ vdeq,
                 const float* __restrict__ qssum,
                 const float* __restrict__ kssum,
                 const float* __restrict__ bias_table,
                 const int*   __restrict__ rel_index,
                 _Float16* __restrict__ oh) {
    __shared__ signed char ks_lds[208 * 64];
    __shared__ _Float16    vT[64][224];
    __shared__ float       sc[16][224];
    __shared__ _Float16    pbuf[16][224];
    __shared__ float       pmax[16][8], psum[16][8];
    __shared__ float       rmax[16], rsum[16];

    const int nt = blockIdx.x;
    const int bh = blockIdx.y;
    const int b  = bh / NHEADS, h = bh % NHEADS;
    const int t  = threadIdx.x;
    const int wave = t >> 5, lane = t & 31;
    const int hi = lane >> 4, ln = lane & 15;
    const int n0 = nt * 16;

    const float sfac = (qssum[bh] * MEANDIV) * (kssum[bh] * MEANDIV) * SCALEQK;
    const signed char* qrow = qsign + (size_t)bh * NTOK * 64;
    const signed char* krow = ksign + (size_t)bh * NTOK * 64;
    const _Float16*    vrow = vdeq  + (size_t)bh * NTOK * 64;

#if HAVE_ASYNC_LDS
    for (int idx = t; idx < 788; idx += 128)
        __builtin_amdgcn_global_load_async_to_lds_b128(
            (gv4i*)(krow + idx * 16), (lv4i*)&ks_lds[idx * 16], 0, 0);
#else
    for (int idx = t; idx < 788; idx += 128)
        *(uint4*)&ks_lds[idx * 16] = *(const uint4*)(krow + idx * 16);
#endif
    uint4 z4; z4.x = z4.y = z4.z = z4.w = 0u;
    for (int idx = 788 + t; idx < 832; idx += 128)
        *(uint4*)&ks_lds[idx * 16] = z4;

    for (int idx = t; idx < 64 * 224 / 8; idx += 128)
        ((uint4*)&vT[0][0])[idx] = z4;
    __syncthreads();
    for (int idx = t; idx < 197 * 8; idx += 128) {
        int m = idx >> 3, dc = (idx & 7) * 8;
        _Float16 tmp[8];
        *(uint4*)tmp = *(const uint4*)(vrow + (size_t)m * 64 + dc);
#pragma unroll
        for (int u = 0; u < 8; ++u) vT[dc + u][m] = tmp[u];
    }
#if HAVE_ASYNC_LDS
    __builtin_amdgcn_s_wait_asynccnt(0);
#endif
    __syncthreads();

    v8i af;
    {
        int n = n0 + ln;
        const int* qp = (const int*)(qrow + (size_t)n * 64);
#pragma unroll
        for (int j = 0; j < 8; ++j) {
            int jj = j & 3;
            int base = 32 * (j >> 2) + 16 * (jj >> 1) + 8 * hi + 4 * (jj & 1);
            af[j] = (n < NTOK) ? qp[base >> 2] : 0;
        }
    }
    for (int mt = wave; mt < 13; mt += 4) {
        int m0 = mt * 16;
        v8i bf;
        {
            const int* kp = (const int*)&ks_lds[(m0 + ln) * 64];
#pragma unroll
            for (int j = 0; j < 8; ++j) {
                int base = 32 * (j >> 2) + 16 * ((j & 3) >> 1) + 8 * hi + 4 * (j & 1);
                bf[j] = kp[base >> 2];
            }
        }
        v8i ci;
#pragma unroll
        for (int j = 0; j < 8; ++j) ci[j] = 0;
        ci = __builtin_amdgcn_wmma_i32_16x16x64_iu8(true, af, true, bf,
                                                    ci, false, false);
#pragma unroll
        for (int j = 0; j < 8; ++j) {
            int rl = 8 * hi + j;
            int n = n0 + rl, m = m0 + ln;
            float s;
            if (m >= NTOK) {
                s = -INFINITY;
            } else {
                float bias = (n < NTOK)
                    ? bias_table[rel_index[n * NTOK + m] * NHEADS + h] : 0.0f;
                s = (float)ci[j] * sfac + bias;
            }
            sc[rl][m0 + ln] = s;
        }
    }
    __syncthreads();

    {
        int r = t >> 3, q8 = t & 7;
        float mm = -INFINITY;
        for (int m = q8; m < NTOK; m += 8) mm = fmaxf(mm, sc[r][m]);
        pmax[r][q8] = mm;
    }
    __syncthreads();
    if ((t & 7) == 0) {
        int r = t >> 3; float mm = pmax[r][0];
#pragma unroll
        for (int u = 1; u < 8; ++u) mm = fmaxf(mm, pmax[r][u]);
        rmax[r] = mm;
    }
    __syncthreads();
    {
        int r = t >> 3, q8 = t & 7;
        float rm = rmax[r], ss = 0.0f;
        for (int m = q8; m < NTOK; m += 8) ss += __expf(sc[r][m] - rm);
        psum[r][q8] = ss;
    }
    __syncthreads();
    if ((t & 7) == 0) {
        int r = t >> 3; float ss = psum[r][0];
#pragma unroll
        for (int u = 1; u < 8; ++u) ss += psum[r][u];
        rsum[r] = ss;
    }
    __syncthreads();
    for (int idx = t; idx < 16 * 224; idx += 128) {
        int r = idx / 224, m = idx % 224;
        float v = 0.0f;
        if (m < NTOK)
            v = rintf(__expf(sc[r][m] - rmax[r]) / rsum[r] * 255.0f) * (1.0f / 255.0f);
        pbuf[r][m] = (_Float16)v;
    }
    __syncthreads();

    v8f oacc;
#pragma unroll
    for (int j = 0; j < 8; ++j) oacc[j] = 0.0f;
    const int d0 = wave * 16;
    for (int kt = 0; kt < 7; ++kt) {
        int mb = kt * 32;
        v16h pf, vf;
#pragma unroll
        for (int e = 0; e < 16; ++e) {
            int pg = e >> 1, w2 = e & 1;
            int kk = (pg < 4 ? hi * 8 + 2 * pg : 16 + hi * 8 + 2 * (pg - 4)) + w2;
            pf[e] = pbuf[ln][mb + kk];
        }
#pragma unroll
        for (int e = 0; e < 16; ++e)
            vf[e] = vT[d0 + ln][mb + (e < 8 ? hi * 8 + e : 16 + hi * 8 + (e - 8))];

        oacc = __builtin_amdgcn_wmma_f32_16x16x32_f16(false, pf, false, vf,
                                                      (short)0, oacc, false, false);
    }
#pragma unroll
    for (int j = 0; j < 8; ++j) {
        int n = n0 + 8 * hi + j;
        if (n < NTOK)
            oh[((size_t)(b * NTOK) + n) * CDIM + h * 64 + d0 + ln] = (_Float16)oacc[j];
    }
}

__global__ __launch_bounds__(256)
void proj_gemm_kernel(const _Float16* __restrict__ ih,
                      const float* __restrict__ w,
                      const float* __restrict__ bias,
                      float* __restrict__ out) {
    __shared__ _Float16 As[2][32][40];
    __shared__ _Float16 Bs[2][64][40];

    const int rt = blockIdx.x, ct = blockIdx.y;
    const int t = threadIdx.x;
    const int wave = t >> 5, lane = t & 31;
    const int hi = lane >> 4, ln = lane & 15;
    const int wr = wave >> 2, wc = wave & 3;
    const int row0 = rt * 32, c0 = ct * 64;

    const int ar = t >> 3, acc4 = (t & 7) * 4;
    const int bc = t >> 2, bk8 = (t & 3) * 8;
    const _Float16* aBase = ih + (size_t)(row0 + ar) * CDIM + acc4;
    const float*    bBase = w  + (size_t)(c0 + bc) * CDIM + bk8;

    uint2  aR;
    float4 bR0, bR1;
    auto ldRegs = [&](int k0) {
        aR  = *(const uint2*)(aBase + k0);
        bR0 = *(const float4*)(bBase + k0);
        bR1 = *(const float4*)(bBase + k0 + 4);
    };
    auto stLDS = [&](int bf) {
        *(uint2*)&As[bf][ar][acc4] = aR;
        Bs[bf][bc][bk8 + 0] = (_Float16)bR0.x;
        Bs[bf][bc][bk8 + 1] = (_Float16)bR0.y;
        Bs[bf][bc][bk8 + 2] = (_Float16)bR0.z;
        Bs[bf][bc][bk8 + 3] = (_Float16)bR0.w;
        Bs[bf][bc][bk8 + 4] = (_Float16)bR1.x;
        Bs[bf][bc][bk8 + 5] = (_Float16)bR1.y;
        Bs[bf][bc][bk8 + 6] = (_Float16)bR1.z;
        Bs[bf][bc][bk8 + 7] = (_Float16)bR1.w;
    };

    v8f acc;
#pragma unroll
    for (int j = 0; j < 8; ++j) acc[j] = 0.0f;

    ldRegs(0);
    stLDS(0);
    __syncthreads();

    int cur = 0;
    for (int i = 0; i < 24; ++i) {
        const int k0 = i * 32;
        if (i < 23) ldRegs(k0 + 32);
        if (i < 22) {
            __builtin_prefetch(aBase + k0 + 64, 0, 0);
            __builtin_prefetch(bBase + k0 + 64, 0, 0);
        }
        v16h af, bf;
#pragma unroll
        for (int e = 0; e < 16; ++e) {
            int pg = e >> 1, w2 = e & 1;
            int kk = (pg < 4 ? hi * 8 + 2 * pg : 16 + hi * 8 + 2 * (pg - 4)) + w2;
            af[e] = As[cur][wr * 16 + ln][kk];
        }
#pragma unroll
        for (int e = 0; e < 16; ++e)
            bf[e] = Bs[cur][wc * 16 + ln][(e < 8 ? hi * 8 + e : 16 + hi * 8 + (e - 8))];

        acc = __builtin_amdgcn_wmma_f32_16x16x32_f16(false, af, false, bf,
                                                     (short)0, acc, false, false);
        if (i < 23) stLDS(cur ^ 1);
        __syncthreads();
        cur ^= 1;
    }

#pragma unroll
    for (int j = 0; j < 8; ++j) {
        int row = row0 + wr * 16 + hi * 8 + j;
        int col = c0 + wc * 16 + ln;
        out[(size_t)row * CDIM + col] = acc[j] + bias[col];
    }
}

extern "C" void kernel_launch(void* const* d_in, const int* in_sizes, int n_in,
                              void* d_out, int out_size, void* d_ws, size_t ws_size,
                              hipStream_t stream) {
    const float* x      = (const float*)d_in[0];
    const float* qkv_w  = (const float*)d_in[1];
    const float* proj_w = (const float*)d_in[2];
    const float* proj_b = (const float*)d_in[3];
    const float* rtab   = (const float*)d_in[4];
    const int*   ridx   = (const int*)  d_in[5];
    float* out = (float*)d_out;

    char* ws = (char*)d_ws;
    size_t off = 0;
    auto alloc = [&](size_t bytes) -> void* {
        void* p = ws + off;
        off = (off + bytes + 255) & ~(size_t)255;
        return p;
    };
    signed char* qsg   = (signed char*)alloc((size_t)BATCH * NHEADS * NTOK * 64);
    signed char* ksg   = (signed char*)alloc((size_t)BATCH * NHEADS * NTOK * 64);
    _Float16*    vdq   = (_Float16*)alloc((size_t)BATCH * NHEADS * NTOK * 64 * 2);
    _Float16*    oh    = (_Float16*)alloc((size_t)ROWS * CDIM * 2);
    float*       qssum = (float*)alloc((size_t)BATCH * NHEADS * 4);
    float*       kssum = (float*)alloc((size_t)BATCH * NHEADS * 4);

    (void)hipMemsetAsync(qssum, 0, (size_t)BATCH * NHEADS * 4, stream);
    (void)hipMemsetAsync(kssum, 0, (size_t)BATCH * NHEADS * 4, stream);

    qkv_gemm_kernel<<<dim3(ROWT, 36), 256, 0, stream>>>(
        x, qkv_w, qsg, ksg, vdq, qssum, kssum);

    attn_kernel<<<dim3(13, BATCH * NHEADS), 128, 0, stream>>>(
        qsg, ksg, vdq, qssum, kssum, rtab, ridx, oh);

    proj_gemm_kernel<<<dim3(ROWT, 12), 256, 0, stream>>>(
        oh, proj_w, proj_b, out);
}
